// GraphAttentionLayer_17188459119242
// MI455X (gfx1250) — matrix operand order repaired and checked by reading, not compiled, not run on hardware
//
#include <hip/hip_runtime.h>

// GAT attention layer for MI455X (gfx1250, wave32, WMMA).
// B=2, N=2048, D_IN=256, H=8, DH=32.
// Path: f32 inputs -> bf16 WMMA (v_wmma_f32_16x16x32_bf16) with f32 accumulate,
// flash-attention online softmax, adj mask streamed (33.5MB, fits in 192MB L2).

typedef __attribute__((ext_vector_type(16))) __bf16 v16bf;
typedef __attribute__((ext_vector_type(8)))  __bf16 v8bf;
typedef __attribute__((ext_vector_type(8)))  float  v8f;

#define NROWS 2048
#define DIN   256
#define BB    2
#define HH    8
#define DHD   32

__device__ __forceinline__ v8bf ld8(const __bf16* p) { return *(const v8bf*)p; }

__device__ __forceinline__ v16bf pack16(v8bf lo, v8bf hi) {
  v16bf r;
#pragma unroll
  for (int i = 0; i < 8; ++i) { r[i] = lo[i]; r[8 + i] = hi[i]; }
  return r;
}

__device__ __forceinline__ v16bf cvt16(float4 a, float4 b, float4 c, float4 d) {
  v16bf r;
  r[0]  = (__bf16)a.x; r[1]  = (__bf16)a.y; r[2]  = (__bf16)a.z; r[3]  = (__bf16)a.w;
  r[4]  = (__bf16)b.x; r[5]  = (__bf16)b.y; r[6]  = (__bf16)b.z; r[7]  = (__bf16)b.w;
  r[8]  = (__bf16)c.x; r[9]  = (__bf16)c.y; r[10] = (__bf16)c.z; r[11] = (__bf16)c.w;
  r[12] = (__bf16)d.x; r[13] = (__bf16)d.y; r[14] = (__bf16)d.z; r[15] = (__bf16)d.w;
  return r;
}

// ---------------------------------------------------------------------------
// Kernel 1: Q/K/V projections. One wave computes one 16x16 output tile.
// Q,K stored row-major bf16 [B*N, 256]; V stored TRANSPOSED: Vt[b,h,d,n].
// A-operand (x tile 16x32): per-lane straight K: half0 k=0..15, half1 k=16..31.
// B-operand (W^T tile 32x16): per-lane straight K: half0 k=0..15, half1 k=16..31.
// ---------------------------------------------------------------------------
__global__ __launch_bounds__(256) void qkv_proj_kernel(
    const float* __restrict__ x, const float* __restrict__ Wq,
    const float* __restrict__ Wk, const float* __restrict__ Wv,
    __bf16* __restrict__ Qb, __bf16* __restrict__ Kb, __bf16* __restrict__ Vt) {
  const int wid  = (blockIdx.x * blockDim.x + threadIdx.x) >> 5;
  const int lane = threadIdx.x & 31;
  const int hf   = lane >> 4;      // which 16-lane half
  const int l16  = lane & 15;
  const int mat  = wid >> 12;      // 4096 tiles per matrix (256 M-tiles x 16 N-tiles)
  const int t    = wid & 4095;
  const int m0   = (t >> 4) << 4;
  const int n0   = (t & 15) << 4;
  const float* W = (mat == 0) ? Wq : (mat == 1) ? Wk : Wv;

  const float* xr = x + (size_t)(m0 + l16) * DIN;
  const float* wr = W + (size_t)(n0 + l16) * DIN;

  v8f acc = {};
#pragma unroll
  for (int kb = 0; kb < DIN; kb += 32) {
    const float* pa = xr + kb + (hf ? 16 : 0);
    float4 a0 = *(const float4*)(pa);
    float4 a1 = *(const float4*)(pa + 4);
    float4 a2 = *(const float4*)(pa + 8);
    float4 a3 = *(const float4*)(pa + 12);
    const float* pb = wr + kb + (hf ? 16 : 0);
    float4 b0 = *(const float4*)(pb);
    float4 b1 = *(const float4*)(pb + 4);
    float4 b2 = *(const float4*)(pb + 8);
    float4 b3 = *(const float4*)(pb + 12);
    v16bf A  = cvt16(a0, a1, a2, a3);
    v16bf Bm = cvt16(b0, b1, b2, b3);
    acc = __builtin_amdgcn_wmma_f32_16x16x32_bf16(false, A, false, Bm,
                                                  (short)0, acc, false, false);
  }

  // C layout: lane<16 -> rows m0+j, lane>=16 -> rows m0+8+j; col = n0+l16.
  const int col = n0 + l16;
  if (mat < 2) {
    __bf16* dst = (mat == 0) ? Qb : Kb;
#pragma unroll
    for (int j = 0; j < 8; ++j)
      dst[(size_t)(m0 + hf * 8 + j) * DIN + col] = (__bf16)acc[j];
  } else {
    // Transposed store: Vt[((b*H + h)*DH + d)*N + n], 8 consecutive n per lane.
    const int h    = col >> 5, d = col & 31;
    const int bidx = m0 >> 11;               // row-tile's batch
    const int nb   = (m0 & 2047) + hf * 8;
    v8bf vv;
#pragma unroll
    for (int j = 0; j < 8; ++j) vv[j] = (__bf16)acc[j];
    *(v8bf*)(Vt + ((size_t)(bidx * HH + h) * DHD + d) * NROWS + nb) = vv;
  }
}

// ---------------------------------------------------------------------------
// Kernel 2: fused masked attention with online softmax.
// One workgroup = (batch b, 16 query rows). Wave w = head h (8 waves = 8 heads),
// so all heads stream the SAME adj tile (hits L0/L2; whole adj fits in L2).
// Computes S^T = K_blk * Q^T so each attention row is owned by lane pair
// (r, r^16): one shfl_xor(16) per reduction, and P lands in B-operand layout.
// ---------------------------------------------------------------------------
__global__ __launch_bounds__(256) void gat_attn_kernel(
    const float* __restrict__ adj, const __bf16* __restrict__ Qb,
    const __bf16* __restrict__ Kb, const __bf16* __restrict__ Vt,
    __bf16* __restrict__ Ob) {
  const int b    = blockIdx.x >> 7;          // 128 row-tiles per batch
  const int r0   = (blockIdx.x & 127) << 4;
  const int h    = threadIdx.x >> 5;         // wave id = head
  const int lane = threadIdx.x & 31;
  const int hf   = lane >> 4;
  const int l16  = lane & 15;

  const int gr = b * NROWS + r0;

  // Q^T as B operand (straight K=dh): half0 dh=0..15, half1 dh=16..31.
  const __bf16* qp = Qb + (size_t)(gr + l16) * DIN + h * DHD + hf * 16;
  const v16bf QB = pack16(ld8(qp), ld8(qp + 8));

  const float*  adjrow = adj + ((size_t)b * NROWS + r0 + l16) * NROWS;
  const __bf16* krow   = Kb + (size_t)(b * NROWS) * DIN + h * DHD + hf * 16;
  const __bf16* vrow   = Vt + ((size_t)(b * HH + h) * DHD + l16) * NROWS;

  v8f   o0 = {}, o1 = {};
  float m_run = -1e30f, l_run = 0.0f;
  const float sc = 0.17677669529663687f;     // 1/sqrt(DH)

  for (int cb = 0; cb < NROWS; cb += 32) {
    // K A-tiles (M=c 16, K=dh 32; straight per-lane K: half0 dh=0..15, half1 dh=16..31).
    const __bf16* kp0 = krow + (size_t)(cb + l16) * DIN;
    const __bf16* kp1 = kp0 + (size_t)16 * DIN;
    v16bf KA0 = pack16(ld8(kp0), ld8(kp0 + 8));
    v16bf KA1 = pack16(ld8(kp1), ld8(kp1 + 8));

    v8f s0 = {}, s1 = {};
    s0 = __builtin_amdgcn_wmma_f32_16x16x32_bf16(false, KA0, false, QB,
                                                 (short)0, s0, false, false);
    s1 = __builtin_amdgcn_wmma_f32_16x16x32_bf16(false, KA1, false, QB,
                                                 (short)0, s1, false, false);

    // adj mask: lane (hf,l16) owns row r0+l16, cols cb + hf*8 + {0..7, 16..23}.
    const float* ap = adjrow + cb + hf * 8;
    if (cb + 32 < NROWS) __builtin_prefetch(ap + 32, 0, 0);
    float4 k0 = *(const float4*)(ap);
    float4 k1 = *(const float4*)(ap + 4);
    float4 k2 = *(const float4*)(ap + 16);
    float4 k3 = *(const float4*)(ap + 20);
    float mk0[8] = {k0.x, k0.y, k0.z, k0.w, k1.x, k1.y, k1.z, k1.w};
    float mk1[8] = {k2.x, k2.y, k2.z, k2.w, k3.x, k3.y, k3.z, k3.w};

    float bm = -1e30f;
#pragma unroll
    for (int j = 0; j < 8; ++j) {
      s0[j] *= sc; s1[j] *= sc;
      bm = fmaxf(bm, (mk0[j] != 0.0f) ? s0[j] : -1e30f);
      bm = fmaxf(bm, (mk1[j] != 0.0f) ? s1[j] : -1e30f);
    }
    bm = fmaxf(bm, __shfl_xor(bm, 16, 32));
    const float m_new = fmaxf(m_run, bm);
    const float alpha = __expf(m_run - m_new);

    float p0[8], p1[8], ps = 0.0f;
#pragma unroll
    for (int j = 0; j < 8; ++j) {
      p0[j] = (mk0[j] != 0.0f) ? __expf(s0[j] - m_new) : 0.0f;
      p1[j] = (mk1[j] != 0.0f) ? __expf(s1[j] - m_new) : 0.0f;
      ps += p0[j] + p1[j];
    }
    ps += __shfl_xor(ps, 16, 32);
    l_run = l_run * alpha + ps;
    m_run = m_new;
#pragma unroll
    for (int j = 0; j < 8; ++j) { o0[j] *= alpha; o1[j] *= alpha; }

    // P^T as B operand (straight K=c layout): swap 8 values with xor-16 partner.
    v16bf PB;
#pragma unroll
    for (int j = 0; j < 8; ++j) {
      const float q0 = __shfl_xor(p0[j], 16, 32);
      const float q1 = __shfl_xor(p1[j], 16, 32);
      PB[j]     = (__bf16)(hf ? q1    : p0[j]);
      PB[8 + j] = (__bf16)(hf ? p1[j] : q0);
    }

    // V^T A-tiles from Vt (M=d, K=c; contiguous in n -> two b128 loads each).
    const __bf16* vp0 = vrow + cb + hf * 16;
    const __bf16* vp1 = vp0 + (size_t)16 * NROWS;
    v16bf VA0 = pack16(ld8(vp0), ld8(vp0 + 8));
    v16bf VA1 = pack16(ld8(vp1), ld8(vp1 + 8));

    o0 = __builtin_amdgcn_wmma_f32_16x16x32_bf16(false, VA0, false, PB,
                                                 (short)0, o0, false, false);
    o1 = __builtin_amdgcn_wmma_f32_16x16x32_bf16(false, VA1, false, PB,
                                                 (short)0, o1, false, false);
  }

  // Self-loop in adj guarantees l_run > 0.
  const float inv = 1.0f / l_run;
  __bf16* op = Ob + (size_t)(gr + l16) * DIN + h * DHD + hf * 8;
  v8bf w0, w1;
#pragma unroll
  for (int j = 0; j < 8; ++j) {
    w0[j] = (__bf16)(o0[j] * inv);
    w1[j] = (__bf16)(o1[j] * inv);
  }
  *(v8bf*)op        = w0;   // d = hf*8 + j
  *(v8bf*)(op + 16) = w1;   // d = 16 + hf*8 + j
}

// ---------------------------------------------------------------------------
// Kernel 3: out = attn @ Wo^T + bo, f32 output. One wave per 16x16 tile.
// ---------------------------------------------------------------------------
__global__ __launch_bounds__(256) void out_proj_kernel(
    const __bf16* __restrict__ Ob, const float* __restrict__ Wo,
    const float* __restrict__ bo, float* __restrict__ out) {
  const int wid  = (blockIdx.x * blockDim.x + threadIdx.x) >> 5;
  const int lane = threadIdx.x & 31;
  const int hf   = lane >> 4;
  const int l16  = lane & 15;
  const int m0   = (wid >> 4) << 4;
  const int n0   = (wid & 15) << 4;

  const __bf16* ar = Ob + (size_t)(m0 + l16) * DIN + hf * 16;
  const float*  wr = Wo + (size_t)(n0 + l16) * DIN + hf * 16;

  v8f acc = {};
#pragma unroll
  for (int kb = 0; kb < DIN; kb += 32) {
    v16bf A = pack16(ld8(ar + kb), ld8(ar + kb + 8));
    const float* pb = wr + kb;
    float4 b0 = *(const float4*)(pb);
    float4 b1 = *(const float4*)(pb + 4);
    float4 b2 = *(const float4*)(pb + 8);
    float4 b3 = *(const float4*)(pb + 12);
    v16bf Bm = cvt16(b0, b1, b2, b3);
    acc = __builtin_amdgcn_wmma_f32_16x16x32_bf16(false, A, false, Bm,
                                                  (short)0, acc, false, false);
  }
  const int   col  = n0 + l16;
  const float bias = bo[col];
#pragma unroll
  for (int j = 0; j < 8; ++j)
    out[(size_t)(m0 + hf * 8 + j) * DIN + col] = acc[j] + bias;
}

// ---------------------------------------------------------------------------
extern "C" void kernel_launch(void* const* d_in, const int* in_sizes, int n_in,
                              void* d_out, int out_size, void* d_ws, size_t ws_size,
                              hipStream_t stream) {
  const float* x   = (const float*)d_in[0];
  const float* adj = (const float*)d_in[1];
  const float* Wq  = (const float*)d_in[2];
  const float* Wk  = (const float*)d_in[3];
  const float* Wv  = (const float*)d_in[4];
  const float* Wo  = (const float*)d_in[5];
  const float* bo  = (const float*)d_in[6];
  float* out = (float*)d_out;

  const size_t BN = (size_t)BB * NROWS;          // 4096 rows
  __bf16* Qb = (__bf16*)d_ws;                    // [BN,256] bf16   2 MB
  __bf16* Kb = Qb + BN * DIN;                    // [BN,256] bf16   2 MB
  __bf16* Vt = Kb + BN * DIN;                    // [B,H,DH,N] bf16 2 MB
  __bf16* Ob = Vt + BN * DIN;                    // [BN,256] bf16   2 MB

  // 3 matrices * 4096 tiles, 8 waves/block -> 1536 blocks.
  qkv_proj_kernel<<<1536, 256, 0, stream>>>(x, Wq, Wk, Wv, Qb, Kb, Vt);
  // B * (N/16) = 256 blocks; 8 waves = 8 heads per block.
  gat_attn_kernel<<<256, 256, 0, stream>>>(adj, Qb, Kb, Vt, Ob);
  // 4096 tiles -> 512 blocks.
  out_proj_kernel<<<512, 256, 0, stream>>>(Ob, Wo, bo, out);
}
